// XRDMamba_67611375173788
// MI455X (gfx1250) — compile-verified, matrix operand order checked by reading, not run on hardware
//
#include <hip/hip_runtime.h>
#include <hip/hip_bf16.h>
#include <math.h>

typedef __attribute__((ext_vector_type(16))) _Float16 v16h;
typedef __attribute__((ext_vector_type(8)))  float    v8f;

#define B_   128
#define L_   5000
#define DM   8      // d_model
#define NL   4      // layers
#define DS   16     // d_state
#define DI   16     // d_inner
#define DC   3      // conv kernel
#define HID_ 64
#define NC   230

// ---------------------------------------------------------------- embed * x
__global__ __launch_bounds__(256) void embed_mul_kernel(
    const float* __restrict__ x, const int* __restrict__ idx,
    const float* __restrict__ embed, float* __restrict__ f)
{
    int i = blockIdx.x * 256 + threadIdx.x;
    const int total = B_ * L_ * DM;
    if (i >= total) return;
    int d   = i & (DM - 1);
    int bt  = i / DM;
    int t   = bt % L_;
    int b   = bt / L_;
    f[i] = embed[idx[t] * DM + d] * x[b * L_ + t];
}

// ---------------------------------------------------------------- zero util
__global__ __launch_bounds__(256) void zero_kernel(float* __restrict__ p, int n)
{
    int i = blockIdx.x * 256 + threadIdx.x;
    if (i < n) p[i] = 0.f;
}

__device__ __forceinline__ float silu_f(float v) {
    return v * (1.f / (1.f + __expf(-v)));
}

// ---------------------------------------------------------------- mamba layer
// One wave (32 lanes) per batch row. Sequential scan over L.
// Lane roles: lane i<32 owns in_proj row i (lanes 0..15 -> xc channel, 16..31 -> z).
// SSM state: lane l owns e = l>>1, n in [8*(l&1), +8): 8 states/lane in VGPRs.
__global__ __launch_bounds__(32) void mamba_layer_kernel(
    float* __restrict__ f,
    const float* __restrict__ norm_w,   const float* __restrict__ in_proj_w,
    const float* __restrict__ conv_w,   const float* __restrict__ conv_b,
    const float* __restrict__ x_proj_w, const float* __restrict__ dt_proj_w,
    const float* __restrict__ dt_proj_b,const float* __restrict__ A_log,
    const float* __restrict__ Dp,       const float* __restrict__ out_proj_w,
    int layer)
{
    const int b    = blockIdx.x;
    const int lane = threadIdx.x;

    // ---- per-lane constant preload (all in VGPRs)
    float nw[DM], wip[DM];
#pragma unroll
    for (int d = 0; d < DM; ++d) {
        nw[d]  = norm_w[layer * DM + d];
        wip[d] = in_proj_w[(layer * 2 * DI + lane) * DM + d];
    }
    float xp0[DI], xpR[DI];
#pragma unroll
    for (int q = 0; q < DI; ++q) {
        xp0[q] = x_proj_w[(layer * (1 + 2 * DS) + 0) * DI + q];            // dt row
        xpR[q] = x_proj_w[(layer * (1 + 2 * DS) + 1 + lane) * DI + q];     // B/C row
    }
    const int ch  = lane & 15;       // depthwise-conv channel (lanes 0..15)
    const float cw0 = conv_w[(layer * DI + ch) * DC + 0];
    const float cw1 = conv_w[(layer * DI + ch) * DC + 1];
    const float cw2 = conv_w[(layer * DI + ch) * DC + 2];
    const float cb  = conv_b[layer * DI + ch];

    const int e  = lane >> 1;
    const int n0 = (lane & 1) * 8;
    const float dtw = dt_proj_w[layer * DI + e];
    const float dtb = dt_proj_b[layer * DI + e];
    const float Dpe = Dp[layer * DI + e];
    float Ar[8];
#pragma unroll
    for (int j = 0; j < 8; ++j)
        Ar[j] = -__expf(A_log[(layer * DI + e) * DS + n0 + j]);

    float wout[DI];
#pragma unroll
    for (int q = 0; q < DI; ++q)
        wout[q] = out_proj_w[(layer * DM + (lane & 7)) * DI + q];

    float hst[8];
#pragma unroll
    for (int j = 0; j < 8; ++j) hst[j] = 0.f;
    float c_h1 = 0.f, c_h2 = 0.f;    // causal conv history

    float* frow = f + (size_t)b * L_ * DM;

    for (int t = 0; t < L_; ++t) {
        // all lanes read the 8 model dims (broadcast-friendly 2x float4)
        const float4* fp = (const float4*)(frow + t * DM);
        float4 f0 = fp[0], f1 = fp[1];

        float ss = f0.x*f0.x + f0.y*f0.y + f0.z*f0.z + f0.w*f0.w
                 + f1.x*f1.x + f1.y*f1.y + f1.z*f1.z + f1.w*f1.w;
        float rms = rsqrtf(ss * (1.f / DM) + 1e-5f);
        float xn[DM] = { f0.x, f0.y, f0.z, f0.w, f1.x, f1.y, f1.z, f1.w };
        float xz = 0.f;
#pragma unroll
        for (int d = 0; d < DM; ++d) xz += wip[d] * (xn[d] * rms * nw[d]);

        // depthwise causal conv (meaningful in lanes 0..15) + SiLU
        float xcpre = xz;
        float cv = cw0 * c_h2 + cw1 * c_h1 + cw2 * xcpre + cb;
        c_h2 = c_h1; c_h1 = xcpre;
        float xc = silu_f(cv);         // lanes 0..15: conv output
        float zs = silu_f(xz);         // lanes 16..31: silu(z)

        // x_proj: every lane needs all 16 xc channels
        float dtin = 0.f, prj = 0.f;
#pragma unroll
        for (int q = 0; q < 16; ++q) {
            float xq = __shfl(xc, q, 32);
            dtin += xp0[q] * xq;
            prj  += xpR[q] * xq;
        }
        // prj: lanes 0..15 hold Bm[lane], lanes 16..31 hold Cm[lane-16]
        float Bn[8], Cn[8];
#pragma unroll
        for (int j = 0; j < 8; ++j) {
            Bn[j] = __shfl(prj, n0 + j, 32);
            Cn[j] = __shfl(prj, 16 + n0 + j, 32);
        }
        float xdt   = dtin * dtw + dtb;
        float delta = (xdt > 20.f) ? xdt : log1pf(__expf(xdt));
        float u     = __shfl(xc, e, 32);
        float du    = delta * u;

        float yp = 0.f;
#pragma unroll
        for (int j = 0; j < 8; ++j) {
            float dA = __expf(delta * Ar[j]);
            hst[j] = dA * hst[j] + du * Bn[j];
            yp += hst[j] * Cn[j];
        }
        float y   = yp + __shfl_xor(yp, 1, 32);         // sum over n halves
        float zse = __shfl(zs, 16 + e, 32);
        float g   = (y + Dpe * u) * zse;                // gated output, per e (pair lanes)

        float outd = 0.f;
#pragma unroll
        for (int q = 0; q < 16; ++q) outd += __shfl(g, 2 * q, 32) * wout[q];

        if (lane < 8) {
            float fold = (lane & 4) ? ((lane & 2) ? ((lane & 1) ? f1.w : f1.z)
                                                  : ((lane & 1) ? f1.y : f1.x))
                                    : ((lane & 2) ? ((lane & 1) ? f0.w : f0.z)
                                                  : ((lane & 1) ? f0.y : f0.x));
            frow[t * DM + lane] = fold + outd;          // residual
        }
    }
}

// ---------------------------------------------------------------- classifier
// Per block: b = blockIdx.y, 128 output positions starting at t0.
//   h over [t0-18, t0+146)  (164 pos, f16 in LDS)
//   r = relu(conv2(h)+b2) over [t0-16, t0+144) (160 pos, 10 n-tiles, WMMA)
//   out = relu(h + conv3(r)+b3) over [t0, t0+128) (8 n-tiles, WMMA) -> pooled
#define TILE  128
#define HLEN  164
#define FLEN  166
#define RLEN  160

__device__ __forceinline__ int wmma_k_of(int j, int hi) {
    return (j < 8) ? (8 * hi + j) : (16 + 8 * hi + (j - 8));
}

__global__ __launch_bounds__(256) void classifier_kernel(
    const float* __restrict__ f,
    const float* __restrict__ c1_w, const float* __restrict__ c1_b,
    const float* __restrict__ c2_w, const float* __restrict__ c2_b,
    const float* __restrict__ c3_w, const float* __restrict__ c3_b,
    float* __restrict__ pooled)
{
    __shared__ float    f_s[FLEN * DM];
    __shared__ _Float16 h_s[HLEN * HID_];
    __shared__ _Float16 r_s[RLEN * HID_];
    __shared__ float    c1w_s[HID_ * DM * DC];
    __shared__ float    c1b_s[HID_];

    const int b   = blockIdx.y;
    const int t0  = blockIdx.x * TILE;
    const int tid = threadIdx.x;

    for (int i = tid; i < HID_ * DM * DC; i += 256) c1w_s[i] = c1_w[i];
    if (tid < HID_) c1b_s[tid] = c1_b[tid];

    const float* frow = f + (size_t)b * L_ * DM;
    for (int i = tid; i < FLEN * DM; i += 256) {
        int p = i / DM, d = i % DM;
        int t = t0 - 19 + p;
        f_s[i] = (t >= 0 && t < L_) ? frow[t * DM + d] : 0.f;
    }
    __syncthreads();

    // conv1 + relu -> h (zero outside [0,L): conv2 sees zero padding)
    for (int i = tid; i < HLEN * HID_; i += 256) {
        int p = i / HID_, c = i % HID_;
        int t = t0 - 18 + p;
        float acc = c1b_s[c];
#pragma unroll
        for (int k = 0; k < DC; ++k)
#pragma unroll
            for (int d = 0; d < DM; ++d)
                acc += c1w_s[c * (DM * DC) + d * DC + k] * f_s[(p + k) * DM + d];
        float hv = fmaxf(acc, 0.f);
        h_s[p * HID_ + c] = (_Float16)((t >= 0 && t < L_) ? hv : 0.f);
    }
    __syncthreads();

    const int w    = tid >> 5, lane = tid & 31;
    const int mt   = w & 3;            // output-channel tile (4 x 16)
    const int half = w >> 2;           // position-range half
    const int nloc = lane & 15, hi = lane >> 4;

    // ---- conv2: load W2 A-fragments (3 taps x 2 K-chunks) for this m-tile
    v16h a2[3][2];
#pragma unroll
    for (int k = 0; k < 3; ++k)
#pragma unroll
        for (int q = 0; q < 2; ++q) {
            v16h a = {};
#pragma unroll
            for (int j = 0; j < 16; ++j) {
                int cin = 32 * q + wmma_k_of(j, hi);
                int m   = 16 * mt + nloc;
                a[j] = (_Float16)c2_w[(m * HID_ + cin) * DC + k];
            }
            a2[k][q] = a;
        }
    float bias2[8];
#pragma unroll
    for (int v = 0; v < 8; ++v) bias2[v] = c2_b[16 * mt + v + 8 * hi];

    for (int nt = half * 5; nt < half * 5 + 5; ++nt) {
        v8f acc = {};
#pragma unroll
        for (int k = 0; k < 3; ++k)
#pragma unroll
            for (int q = 0; q < 2; ++q) {
                int prow = 16 * nt + nloc + k + 1;       // h_s row
                v16h bf = {};
#pragma unroll
                for (int j = 0; j < 16; ++j)
                    bf[j] = h_s[prow * HID_ + 32 * q + wmma_k_of(j, hi)];
                acc = __builtin_amdgcn_wmma_f32_16x16x32_f16(
                        false, a2[k][q], false, bf, (short)0, acc, false, false);
            }
        int pidx = 16 * nt + nloc;                       // r index 0..159
        int tg   = t0 - 16 + pidx;                       // global position
#pragma unroll
        for (int v = 0; v < 8; ++v) {
            float rv = fmaxf(acc[v] + bias2[v], 0.f);
            if (tg < 0 || tg >= L_) rv = 0.f;            // conv3 zero padding
            r_s[pidx * HID_ + 16 * mt + v + 8 * hi] = (_Float16)rv;
        }
    }
    __syncthreads();

    // ---- conv3 + residual + relu + pool
    v16h a3[3][2];
#pragma unroll
    for (int k = 0; k < 3; ++k)
#pragma unroll
        for (int q = 0; q < 2; ++q) {
            v16h a = {};
#pragma unroll
            for (int j = 0; j < 16; ++j) {
                int cin = 32 * q + wmma_k_of(j, hi);
                int m   = 16 * mt + nloc;
                a[j] = (_Float16)c3_w[(m * HID_ + cin) * DC + k];
            }
            a3[k][q] = a;
        }
    float bias3[8];
#pragma unroll
    for (int v = 0; v < 8; ++v) bias3[v] = c3_b[16 * mt + v + 8 * hi];

    float psum[8];
#pragma unroll
    for (int v = 0; v < 8; ++v) psum[v] = 0.f;

    for (int nt = half * 4; nt < half * 4 + 4; ++nt) {
        v8f acc = {};
#pragma unroll
        for (int k = 0; k < 3; ++k)
#pragma unroll
            for (int q = 0; q < 2; ++q) {
                int prow = 16 * nt + nloc + k + 15;      // r_s row
                v16h bf = {};
#pragma unroll
                for (int j = 0; j < 16; ++j)
                    bf[j] = r_s[prow * HID_ + 32 * q + wmma_k_of(j, hi)];
                acc = __builtin_amdgcn_wmma_f32_16x16x32_f16(
                        false, a3[k][q], false, bf, (short)0, acc, false, false);
            }
        int pidx = 16 * nt + nloc;                       // 0..127 rel t0
        int tg   = t0 + pidx;
        int hrow = pidx + 18;
#pragma unroll
        for (int v = 0; v < 8; ++v) {
            float hval = (float)h_s[hrow * HID_ + 16 * mt + v + 8 * hi];
            float val  = fmaxf(hval + acc[v] + bias3[v], 0.f);
            if (tg < L_) psum[v] += val;
        }
    }
    // reduce over the 16 positions (n) held across lanes of each half
#pragma unroll
    for (int v = 0; v < 8; ++v) {
#pragma unroll
        for (int off = 1; off < 16; off <<= 1)
            psum[v] += __shfl_xor(psum[v], off, 32);
    }
    if (nloc == 0) {
#pragma unroll
        for (int v = 0; v < 8; ++v)
            atomicAdd(&pooled[b * HID_ + 16 * mt + v + 8 * hi], psum[v]);
    }
}

// ---------------------------------------------------------------- FC head
__global__ __launch_bounds__(256) void fc_kernel(
    const float* __restrict__ pooled, const float* __restrict__ fc_w,
    const float* __restrict__ fc_b, float* __restrict__ out)
{
    int i = blockIdx.x * 256 + threadIdx.x;
    if (i >= B_ * NC) return;
    int b = i / NC, c = i % NC;
    float acc = fc_b[c];
    const float inv_l = 1.f / (float)L_;
#pragma unroll 8
    for (int m = 0; m < HID_; ++m)
        acc += pooled[b * HID_ + m] * inv_l * fc_w[c * HID_ + m];
    out[b * NC + c] = acc;
}

// ---------------------------------------------------------------- launcher
extern "C" void kernel_launch(void* const* d_in, const int* in_sizes, int n_in,
                              void* d_out, int out_size, void* d_ws, size_t ws_size,
                              hipStream_t stream) {
    const float* x         = (const float*)d_in[0];
    const int*   idx       = (const int*)  d_in[1];
    const float* embed     = (const float*)d_in[2];
    const float* norm_w    = (const float*)d_in[3];
    const float* in_proj_w = (const float*)d_in[4];
    const float* conv_w    = (const float*)d_in[5];
    const float* conv_b    = (const float*)d_in[6];
    const float* x_proj_w  = (const float*)d_in[7];
    const float* dt_proj_w = (const float*)d_in[8];
    const float* dt_proj_b = (const float*)d_in[9];
    const float* A_log     = (const float*)d_in[10];
    const float* Dp        = (const float*)d_in[11];
    const float* out_proj_w= (const float*)d_in[12];
    const float* c1_w      = (const float*)d_in[13];
    const float* c1_b      = (const float*)d_in[14];
    const float* c2_w      = (const float*)d_in[15];
    const float* c2_b      = (const float*)d_in[16];
    const float* c3_w      = (const float*)d_in[17];
    const float* c3_b      = (const float*)d_in[18];
    const float* fc_w      = (const float*)d_in[19];
    const float* fc_b      = (const float*)d_in[20];
    float* out = (float*)d_out;

    float* fbuf   = (float*)d_ws;                        // B*L*DM floats (20.48 MB)
    float* pooled = fbuf + (size_t)B_ * L_ * DM;         // B*HID floats

    embed_mul_kernel<<<(B_ * L_ * DM + 255) / 256, 256, 0, stream>>>(x, idx, embed, fbuf);

    for (int l = 0; l < NL; ++l)
        mamba_layer_kernel<<<B_, 32, 0, stream>>>(
            fbuf, norm_w, in_proj_w, conv_w, conv_b, x_proj_w,
            dt_proj_w, dt_proj_b, A_log, Dp, out_proj_w, l);

    zero_kernel<<<(B_ * HID_ + 255) / 256, 256, 0, stream>>>(pooled, B_ * HID_);

    classifier_kernel<<<dim3((L_ + TILE - 1) / TILE, B_), 256, 0, stream>>>(
        fbuf, c1_w, c1_b, c2_w, c2_b, c3_w, c3_b, pooled);

    fc_kernel<<<(B_ * NC + 255) / 256, 256, 0, stream>>>(pooled, fc_w, fc_b, out);
}
